// LSTMModel_58256936403638
// MI455X (gfx1250) — matrix operand order repaired and checked by reading, not compiled, not run on hardware
//
#include <hip/hip_runtime.h>

// ---------------------------------------------------------------------------
// 2-layer LSTM (B=256, T=512, I=128, H=512) + FC head for MI455X (gfx1250).
// Strategy: bf16 WMMA (v_wmma_f32_16x16x32_bf16) with f32 accumulation.
// Recurrence parallelized over batch: 16 workgroups x 16-row batch tiles,
// each iterates all T steps internally (no inter-workgroup sync).
// Weights packed once as bf16 [Wih | Whh] (L2-resident, ~6.5MB total).
// ---------------------------------------------------------------------------

typedef __attribute__((ext_vector_type(16))) __bf16 v16bf;
typedef __attribute__((ext_vector_type(8)))  __bf16 v8bf;
typedef __attribute__((ext_vector_type(4)))  __bf16 v4bf;
typedef __attribute__((ext_vector_type(8)))  float  v8f;

#define B_  256
#define T_  512
#define I_  128
#define H_  512
#define NG  2048   // 4*H gate width

__device__ __forceinline__ float sigmoid_f(float x) {
  return 1.f / (1.f + __expf(-x));
}
__device__ __forceinline__ float tanh_f(float x) {
  float e = __expf(-2.f * fabsf(x));
  float r = (1.f - e) / (1.f + e);
  return copysignf(r, x);
}

// Pack [Wih | Whh] (f32, row-major [NG x KX], [NG x H]) -> bf16 [NG x (KX+H)].
__global__ void pack_weights(const float* __restrict__ Wih,
                             const float* __restrict__ Whh,
                             __bf16* __restrict__ dst, int KX) {
  int Kc = KX + H_;
  int total = NG * Kc;
  for (int i = blockIdx.x * blockDim.x + threadIdx.x; i < total;
       i += gridDim.x * blockDim.x) {
    int n = i / Kc, k = i - n * Kc;
    float v = (k < KX) ? Wih[n * KX + k] : Whh[n * H_ + (k - KX)];
    dst[i] = (__bf16)v;
  }
}

__global__ void sum_bias(const float* __restrict__ a,
                         const float* __restrict__ b,
                         float* __restrict__ o) {
  int i = blockIdx.x * blockDim.x + threadIdx.x;
  if (i < NG) o[i] = a[i] + b[i];
}

// ---------------------------------------------------------------------------
// One LSTM layer. Workgroup = 16 batch rows x full 2048 gates, 16 waves.
// Wave w owns h-columns [32w, 32w+32): gate tiles {i,f,g,o} x {2w, 2w+1}.
// A operand [16 x Kc] bf16 staged in LDS: cols [0,KX) = x_t, [KX,Kc) = h_{t-1}.
// B operand streamed from packed bf16 weights (L2-resident).
//
// Per-lane operand layouts (wave32, ln = lane&15, hi = lane>>4):
//   A frag: row m=ln, halves [kt*32 + 16*hi .. +16)
//   B frag: W row n (gate col), halves [kt*32 + 16*hi .. +16)  (32B load)
//   C/D   : col n=ln, rows m = 8*hi + vgpr
// ---------------------------------------------------------------------------
template <int KX, bool XBF16, bool WRITE_SEQ>
__global__ __launch_bounds__(512) void lstm_layer(
    const float*  __restrict__ xf,     // layer0 input (f32)  [B][T][KX]
    const __bf16* __restrict__ xb,     // layer1 input (bf16) [B][T][KX]
    const __bf16* __restrict__ Wcat,   // [NG][KX+H] bf16 row-major
    const float*  __restrict__ bias,   // [NG] fused b_ih + b_hh
    __bf16* __restrict__ hseq,         // WRITE_SEQ:  [B][T][H] bf16 out
    __bf16* __restrict__ hlast)        // !WRITE_SEQ: [B][H] bf16 out
{
  constexpr int Kc   = KX + H_;
  constexpr int LDSK = Kc + 8;           // +8 halves pad: row stride 516 dwords
  __shared__ __bf16 Ald[16 * LDSK];      // -> 16 A-rows hit distinct LDS banks

  const int tid  = threadIdx.x;
  const int wave = tid >> 5;
  const int lane = tid & 31;
  const int ln   = lane & 15;
  const int hi   = lane >> 4;
  const int bm0  = blockIdx.x * 16;      // first batch row of this tile

  // h0 == 0
  for (int i = tid; i < 16 * H_; i += 512)
    Ald[(i >> 9) * LDSK + KX + (i & (H_ - 1))] = (__bf16)0.f;

  // Per-lane bias values + weight-row pointers for the 8 owned gate tiles.
  // nt = chunk*2 + p,  chunk: 0=i 1=f 2=g 3=o,  tile = chunk*32 + wave*2 + p.
  float bias_v[8];
  const __bf16* pB[8];
#pragma unroll
  for (int nt = 0; nt < 8; ++nt) {
    int chunk = nt >> 1, p = nt & 1;
    int gn = (chunk * 32 + wave * 2 + p) * 16 + ln;
    bias_v[nt] = bias[gn];
    pB[nt] = Wcat + (size_t)gn * Kc + hi * 16;
  }

  const __bf16* pA0 = Ald + ln * LDSK + hi * 16;

  float cst[2][8];                        // cell state, c0 == 0
#pragma unroll
  for (int p = 0; p < 2; ++p)
#pragma unroll
    for (int v = 0; v < 8; ++v) cst[p][v] = 0.f;

#pragma unroll 1
  for (int t = 0; t < T_; ++t) {
    // ---- stage x_t into A[:, 0:KX) ----
    if (XBF16) {
      constexpr int CH = 16 * KX / 8;     // 8 halves per chunk (16B)
      for (int c = tid; c < CH; c += 512) {
        int m = c / (KX / 8), kc = c % (KX / 8);
        *(uint4*)(Ald + m * LDSK + kc * 8) =
            *(const uint4*)(xb + ((size_t)(bm0 + m) * T_ + t) * KX + kc * 8);
      }
    } else {
      constexpr int CH = 16 * KX / 4;     // 4 f32 -> 4 bf16 per chunk
      for (int c = tid; c < CH; c += 512) {
        int m = c / (KX / 4), kc = c % (KX / 4);
        float4 v = *(const float4*)(xf + ((size_t)(bm0 + m) * T_ + t) * KX + kc * 4);
        v4bf o;
        o[0] = (__bf16)v.x; o[1] = (__bf16)v.y;
        o[2] = (__bf16)v.z; o[3] = (__bf16)v.w;
        *(v4bf*)(Ald + m * LDSK + kc * 4) = o;
      }
    }
    __syncthreads();   // x staged; previous step's h writes visible

    // ---- gates = [x_t | h_{t-1}] @ Wcat^T + bias  (WMMA bf16 -> f32) ----
    v8f acc[8];
#pragma unroll
    for (int nt = 0; nt < 8; ++nt)
#pragma unroll
      for (int j = 0; j < 8; ++j) acc[nt][j] = bias_v[nt];

#pragma unroll 2
    for (int kt = 0; kt < Kc / 32; ++kt) {
      v8bf alo = *(const v8bf*)(pA0 + kt * 32);
      v8bf ahi = *(const v8bf*)(pA0 + kt * 32 + 8);
      v16bf afrag = __builtin_shufflevector(
          alo, ahi, 0, 1, 2, 3, 4, 5, 6, 7, 8, 9, 10, 11, 12, 13, 14, 15);
#pragma unroll
      for (int nt = 0; nt < 8; ++nt) {
        v16bf bfrag = *(const v16bf*)(pB[nt] + (size_t)kt * 32);
        acc[nt] = __builtin_amdgcn_wmma_f32_16x16x32_bf16(
            false, afrag, false, bfrag, (short)0, acc[nt], false, false);
      }
    }
    __syncthreads();   // all A reads complete before overwriting h region

    // ---- pointwise cell update for this wave's 32 h-columns ----
#pragma unroll
    for (int p = 0; p < 2; ++p) {
#pragma unroll
      for (int v = 0; v < 8; ++v) {
        float i_ = sigmoid_f(acc[0 + p][v]);
        float f_ = sigmoid_f(acc[2 + p][v]);
        float g_ = tanh_f  (acc[4 + p][v]);
        float o_ = sigmoid_f(acc[6 + p][v]);
        float c  = f_ * cst[p][v] + i_ * g_;
        cst[p][v] = c;
        float h  = o_ * tanh_f(c);
        int m = hi * 8 + v;
        int j = (wave * 2 + p) * 16 + ln;
        __bf16 hb = (__bf16)h;
        Ald[m * LDSK + KX + j] = hb;                       // feed next step
        if (WRITE_SEQ)
          hseq[((size_t)(bm0 + m) * T_ + t) * H_ + j] = hb;
      }
    }
  }

  if (!WRITE_SEQ) {    // emit final hidden state only
    __syncthreads();
    for (int i = tid; i < 16 * H_; i += 512) {
      int m = i >> 9, k = i & (H_ - 1);
      hlast[(size_t)(bm0 + m) * H_ + k] = Ald[m * LDSK + KX + k];
    }
  }
}

// out[b] = dot(h2_last[b,:], Wfc) + bfc — one wave32 per batch row.
__global__ void fc_kernel(const __bf16* __restrict__ h2,
                          const float* __restrict__ Wfc,
                          const float* __restrict__ bfc,
                          float* __restrict__ out) {
  int b = blockIdx.x;
  int lane = threadIdx.x;
  float s = 0.f;
  for (int k = lane; k < H_; k += 32)
    s += (float)h2[(size_t)b * H_ + k] * Wfc[k];
#pragma unroll
  for (int m = 16; m >= 1; m >>= 1) s += __shfl_xor(s, m, 32);
  if (lane == 0) out[b] = s + bfc[0];
}

extern "C" void kernel_launch(void* const* d_in, const int* in_sizes, int n_in,
                              void* d_out, int out_size, void* d_ws, size_t ws_size,
                              hipStream_t stream) {
  (void)in_sizes; (void)n_in; (void)out_size; (void)ws_size;

  const float* x    = (const float*)d_in[0];
  const float* Wih0 = (const float*)d_in[1];
  const float* Whh0 = (const float*)d_in[2];
  const float* bih0 = (const float*)d_in[3];
  const float* bhh0 = (const float*)d_in[4];
  const float* Wih1 = (const float*)d_in[5];
  const float* Whh1 = (const float*)d_in[6];
  const float* bih1 = (const float*)d_in[7];
  const float* bhh1 = (const float*)d_in[8];
  const float* Wfc  = (const float*)d_in[9];
  const float* bfc  = (const float*)d_in[10];
  float* out = (float*)d_out;

  // Workspace layout (~135 MB, all offsets 256B-aligned):
  char* ws = (char*)d_ws;
  size_t off = 0;
  __bf16* Wcat0 = (__bf16*)(ws + off); off += (size_t)NG * (I_ + H_) * 2;  // 2.5MB
  __bf16* Wcat1 = (__bf16*)(ws + off); off += (size_t)NG * (H_ + H_) * 2;  // 4MB
  float*  bias0 = (float*)(ws + off);  off += (size_t)NG * 4;
  float*  bias1 = (float*)(ws + off);  off += (size_t)NG * 4;
  __bf16* h1    = (__bf16*)(ws + off); off += (size_t)B_ * T_ * H_ * 2;    // 128MB
  __bf16* h2l   = (__bf16*)(ws + off); off += (size_t)B_ * H_ * 2;

  // Weight/bias prep (run every call; deterministic, cheap).
  pack_weights<<<1024, 256, 0, stream>>>(Wih0, Whh0, Wcat0, I_);
  pack_weights<<<1024, 256, 0, stream>>>(Wih1, Whh1, Wcat1, H_);
  sum_bias<<<NG / 256, 256, 0, stream>>>(bih0, bhh0, bias0);
  sum_bias<<<NG / 256, 256, 0, stream>>>(bih1, bhh1, bias1);

  // Layer 0: f32 x input, emits bf16 h1 sequence.
  lstm_layer<I_, false, true><<<B_ / 16, 512, 0, stream>>>(
      x, nullptr, Wcat0, bias0, h1, nullptr);
  // Layer 1: bf16 h1 input, emits final hidden state only.
  lstm_layer<H_, true, false><<<B_ / 16, 512, 0, stream>>>(
      nullptr, h1, Wcat1, bias1, nullptr, h2l);

  fc_kernel<<<B_, 32, 0, stream>>>(h2l, Wfc, bfc, out);
}
